// ModalEncoder_28089086116175
// MI455X (gfx1250) — compile-verified, matrix operand order checked by reading, not run on hardware
//
#include <hip/hip_runtime.h>
#include <math.h>

// ---------------- problem constants ----------------
#define NN        50000
#define EE        800000
#define ETOT      (EE + NN)      // edges + self loops
#define FDIM      512
#define HID       64
#define HEADS     4
#define HF        (HEADS * HID)  // 256
#define EMB       64
#define NEG_SLOPE 0.2f
#define LN_EPS    1e-5f

typedef __attribute__((ext_vector_type(16))) __bf16 v16bf;
typedef __attribute__((ext_vector_type(8)))  __bf16 v8bf;
typedef __attribute__((ext_vector_type(8)))  float  v8f;

union BF16x16 { v16bf v; v8bf h[2]; };

// ---------------- small utility kernels ----------------
__global__ void cvt_bf16x4(const float* __restrict__ s, __bf16* __restrict__ d, int n4) {
    int i = blockIdx.x * blockDim.x + threadIdx.x;
    if (i < n4) {
        float4 f = ((const float4*)s)[i];
        d[i * 4 + 0] = (__bf16)f.x;
        d[i * 4 + 1] = (__bf16)f.y;
        d[i * 4 + 2] = (__bf16)f.z;
        d[i * 4 + 3] = (__bf16)f.w;
    }
}

// W [K, Nc] f32 row-major -> WT [Nc, K] bf16 row-major
__global__ void cvt_transpose_bf16(const float* __restrict__ W, __bf16* __restrict__ WT,
                                   int K, int Ncols) {
    int i = blockIdx.x * blockDim.x + threadIdx.x;
    if (i < K * Ncols) {
        int k = i / Ncols, n = i - k * Ncols;
        WT[n * K + k] = (__bf16)W[i];
    }
}

__global__ void fillf(float* __restrict__ p, float v, int n) {
    int i = blockIdx.x * blockDim.x + threadIdx.x;
    if (i < n) p[i] = v;
}

// ---------------- WMMA GEMM ----------------
// C[M,Nc] = A[M,K](bf16) * B[K,Nc] (given as BT[Nc,K] bf16) + bias
// grid.x = M/16 row tiles, grid.y = Nc/64 col groups, 4 waves/block (1 16x16 tile each).
// Fragment layout per CDNA5 ISA 7.12.2 (16-bit A 16x32):
//   lanes 0-15  : M=lane,  K {0..7, 16..23}
//   lanes 16-31 : M=lane-16, K {8..15, 24..31}
// B mirrored with N on lanes (BT rows are B columns -> contiguous 16B loads).
__global__ __launch_bounds__(128) void gemm_bf16_wmma(
    const __bf16* __restrict__ A, const __bf16* __restrict__ BT,
    const float* __restrict__ bias, float* __restrict__ C,
    __bf16* __restrict__ Cbf, int K, int Nc)
{
    const int lane = threadIdx.x & 31;
    const int wave = threadIdx.x >> 5;
    const int m0   = blockIdx.x * 16;
    const int n0   = blockIdx.y * 64 + wave * 16;
    const int half = lane >> 4;   // selects K sub-ranges per layout
    const int idx  = lane & 15;   // A row / B column within tile

    const __bf16* arow = A  + (size_t)(m0 + idx) * K + half * 8;
    const __bf16* bcol = BT + (size_t)(n0 + idx) * K + half * 8;

    v8f acc = {};
    for (int k0 = 0; k0 < K; k0 += 32) {
        BF16x16 a, b;
        a.h[0] = *(const v8bf*)(arow + k0);
        a.h[1] = *(const v8bf*)(arow + k0 + 16);
        b.h[0] = *(const v8bf*)(bcol + k0);
        b.h[1] = *(const v8bf*)(bcol + k0 + 16);
        acc = __builtin_amdgcn_wmma_f32_16x16x32_bf16(
            false, a.v, false, b.v, (short)0, acc, false, false);
    }

    const int cn = n0 + idx;
    const float bv = bias ? bias[cn] : 0.0f;
#pragma unroll
    for (int i = 0; i < 8; ++i) {
        int m = m0 + i + half * 8;          // C/D layout: VGPR i -> M=i (lanes<16) / M=8+i
        float v = acc[i] + bv;
        C[(size_t)m * Nc + cn] = v;
        if (Cbf) Cbf[(size_t)m * Nc + cn] = (__bf16)v;
    }
}

// ---------------- attention logits: als/ald [N,H] ----------------
__global__ void attn_logits(const float* __restrict__ h,
                            const float* __restrict__ asrc,
                            const float* __restrict__ adst,
                            float* __restrict__ als, float* __restrict__ ald) {
    int i = blockIdx.x * blockDim.x + threadIdx.x;   // n*HEADS + head
    if (i >= NN * HEADS) return;
    int n = i >> 2, hd = i & 3;
    const float4* hp = (const float4*)(h + (size_t)n * HF + hd * HID);
    const float4* sp = (const float4*)(asrc + hd * HID);
    const float4* dp = (const float4*)(adst + hd * HID);
    float ss = 0.f, sd = 0.f;
#pragma unroll
    for (int j = 0; j < HID / 4; ++j) {
        float4 hv = hp[j], av = sp[j], bv = dp[j];
        ss += hv.x * av.x + hv.y * av.y + hv.z * av.z + hv.w * av.w;
        sd += hv.x * bv.x + hv.y * bv.y + hv.z * bv.z + hv.w * bv.w;
    }
    als[i] = ss;
    ald[i] = sd;
}

// ---------------- edge passes ----------------
__device__ inline void edge_sd(const int* __restrict__ ei, int e, int& s, int& d) {
    if (e < EE) { s = ei[e]; d = ei[EE + e]; }
    else        { s = d = e - EE; }          // self loops appended
}

__device__ inline void atomicMaxF(float* addr, float val) {
    // ordered-int trick: exact float max semantics, native i32/u32 atomics
    if (val >= 0.f) atomicMax((int*)addr, __float_as_int(val));
    else            atomicMin((unsigned int*)addr, __float_as_uint(val));
}

__global__ void edge_max(const int* __restrict__ ei, const float* __restrict__ als,
                         const float* __restrict__ ald, float* __restrict__ m) {
    int e = blockIdx.x * blockDim.x + threadIdx.x;
    if (e >= ETOT) return;
    int s, d; edge_sd(ei, e, s, d);
#pragma unroll
    for (int hd = 0; hd < HEADS; ++hd) {
        float v = als[s * HEADS + hd] + ald[d * HEADS + hd];
        v = v > 0.f ? v : NEG_SLOPE * v;
        atomicMaxF(&m[d * HEADS + hd], v);
    }
}

__global__ void edge_sum(const int* __restrict__ ei, const float* __restrict__ als,
                         const float* __restrict__ ald, const float* __restrict__ m,
                         float* __restrict__ denom) {
    int e = blockIdx.x * blockDim.x + threadIdx.x;
    if (e >= ETOT) return;
    int s, d; edge_sd(ei, e, s, d);
#pragma unroll
    for (int hd = 0; hd < HEADS; ++hd) {
        float v = als[s * HEADS + hd] + ald[d * HEADS + hd];
        v = v > 0.f ? v : NEG_SLOPE * v;
        atomicAdd(&denom[d * HEADS + hd], __expf(v - m[d * HEADS + hd]));
    }
}

// one wave per edge; lane covers 8 of the 256 channels (head = contiguous 64-chunk).
// h gathered in bf16 (16B/lane, one b128) -> halves the dominant gather traffic;
// accumulation stays in f32 atomics.
__global__ __launch_bounds__(256) void edge_agg(
    const int* __restrict__ ei, const float* __restrict__ als,
    const float* __restrict__ ald, const float* __restrict__ m,
    const float* __restrict__ denom, const __bf16* __restrict__ hbf,
    float* __restrict__ out) {
    int wave = threadIdx.x >> 5, lane = threadIdx.x & 31;
    int e = blockIdx.x * 8 + wave;
    if (e >= ETOT) return;
    int s, d; edge_sd(ei, e, s, d);

    float alpha[HEADS];
#pragma unroll
    for (int hd = 0; hd < HEADS; ++hd) {
        float v = als[s * HEADS + hd] + ald[d * HEADS + hd];
        v = v > 0.f ? v : NEG_SLOPE * v;
        alpha[hd] = __expf(v - m[d * HEADS + hd]) / denom[d * HEADS + hd];
    }

    const v8bf* hp = (const v8bf*)(hbf + (size_t)s * HF);
    float* op = out + (size_t)d * HF;
    int c0 = lane * 8;
    int hd = lane >> 3;   // c0/64
    float al = hd == 0 ? alpha[0] : hd == 1 ? alpha[1] : hd == 2 ? alpha[2] : alpha[3];
    v8bf hv = hp[lane];
#pragma unroll
    for (int j = 0; j < 8; ++j)
        atomicAdd(op + c0 + j, (float)hv[j] * al);
}

// ---------------- head-mean + LayerNorm + PReLU + residual ----------------
// one wave32 per node, lane covers channels {lane, lane+32}
__global__ __launch_bounds__(256) void node_finalize(
    const float* __restrict__ acc, const float* __restrict__ bg,
    const float* __restrict__ gamma, const float* __restrict__ beta,
    const float* __restrict__ prelu, float* __restrict__ x,
    __bf16* __restrict__ xbf) {
    int wave = threadIdx.x >> 5, lane = threadIdx.x & 31;
    int n = blockIdx.x * 8 + wave;
    if (n >= NN) return;
    const float* ap = acc + (size_t)n * HF;

    float v[2];
    float sum = 0.f, sumsq = 0.f;
#pragma unroll
    for (int j = 0; j < 2; ++j) {
        int cc = lane + j * 32;
        float t = 0.25f * (ap[cc] + ap[64 + cc] + ap[128 + cc] + ap[192 + cc]) + bg[cc];
        v[j] = t; sum += t; sumsq += t * t;
    }
#pragma unroll
    for (int off = 16; off; off >>= 1) {
        sum   += __shfl_xor(sum, off, 32);
        sumsq += __shfl_xor(sumsq, off, 32);
    }
    float mu  = sum * (1.f / HID);
    float var = sumsq * (1.f / HID) - mu * mu;
    float inv = rsqrtf(var + LN_EPS);
    float aw  = prelu[0];
#pragma unroll
    for (int j = 0; j < 2; ++j) {
        int cc = lane + j * 32;
        float xn = (v[j] - mu) * inv * gamma[cc] + beta[cc];
        xn = xn > 0.f ? xn : aw * xn;
        float nx = x[(size_t)n * HID + cc] + xn;   // residual
        x[(size_t)n * HID + cc] = nx;
        xbf[(size_t)n * HID + cc] = (__bf16)nx;
    }
}

// ---------------- row L2 normalize ----------------
__global__ __launch_bounds__(256) void l2norm(const float* __restrict__ z,
                                              float* __restrict__ out) {
    int wave = threadIdx.x >> 5, lane = threadIdx.x & 31;
    int n = blockIdx.x * 8 + wave;
    if (n >= NN) return;
    float a = z[(size_t)n * EMB + lane];
    float b = z[(size_t)n * EMB + lane + 32];
    float ss = a * a + b * b;
#pragma unroll
    for (int off = 16; off; off >>= 1) ss += __shfl_xor(ss, off, 32);
    float inv = 1.f / fmaxf(sqrtf(ss), 1e-12f);
    out[(size_t)n * EMB + lane]      = a * inv;
    out[(size_t)n * EMB + lane + 32] = b * inv;
}

// ---------------- host orchestration ----------------
extern "C" void kernel_launch(void* const* d_in, const int* in_sizes, int n_in,
                              void* d_out, int out_size, void* d_ws, size_t ws_size,
                              hipStream_t stream) {
    (void)in_sizes; (void)n_in; (void)out_size; (void)ws_size;

    const float* features = (const float*)d_in[0];
    const int*   ei       = (const int*)d_in[1];
    const float* Wt       = (const float*)d_in[2];
    const float* bt       = (const float*)d_in[3];
    const float* Wp       = (const float*)d_in[4];
    const float* bp       = (const float*)d_in[5];
    const float* Wg[2] = {(const float*)d_in[6],  (const float*)d_in[13]};
    const float* As[2] = {(const float*)d_in[7],  (const float*)d_in[14]};
    const float* Ad[2] = {(const float*)d_in[8],  (const float*)d_in[15]};
    const float* Bg[2] = {(const float*)d_in[9],  (const float*)d_in[16]};
    const float* Gm[2] = {(const float*)d_in[10], (const float*)d_in[17]};
    const float* Bb[2] = {(const float*)d_in[11], (const float*)d_in[18]};
    const float* Pa[2] = {(const float*)d_in[12], (const float*)d_in[19]};

    char* ws = (char*)d_ws;
    size_t off = 0;
    auto carve = [&](size_t bytes) -> char* {
        char* p = ws + off;
        off += (bytes + 255) & ~(size_t)255;
        return p;
    };
    __bf16* feat_bf = (__bf16*)carve((size_t)NN * FDIM * 2);
    __bf16* WtT     = (__bf16*)carve((size_t)FDIM * HID * 2);
    __bf16* WgT0    = (__bf16*)carve((size_t)HID * HF * 2);
    __bf16* WgT1    = (__bf16*)carve((size_t)HID * HF * 2);
    __bf16* WpT     = (__bf16*)carve((size_t)HID * EMB * 2);
    float*  x       = (float*) carve((size_t)NN * HID * 4);
    __bf16* x_bf    = (__bf16*)carve((size_t)NN * HID * 2);
    float*  hbuf    = (float*) carve((size_t)NN * HF * 4);
    __bf16* h_bf    = (__bf16*)carve((size_t)NN * HF * 2);
    float*  als     = (float*) carve((size_t)NN * HEADS * 4);
    float*  ald     = (float*) carve((size_t)NN * HEADS * 4);
    float*  mbuf    = (float*) carve((size_t)NN * HEADS * 4);
    float*  denom   = (float*) carve((size_t)NN * HEADS * 4);
    float*  out_acc = (float*) carve((size_t)NN * HF * 4);
    float*  zbuf    = (float*) carve((size_t)NN * EMB * 4);
    __bf16* WgT[2]  = {WgT0, WgT1};

    const int TB = 256;
    const int MB = NN / 16;          // 3125 row tiles

    // --- weight & feature conversion ---
    cvt_bf16x4<<<(NN * FDIM / 4 + TB - 1) / TB, TB, 0, stream>>>(features, feat_bf, NN * FDIM / 4);
    cvt_transpose_bf16<<<(FDIM * HID + TB - 1) / TB, TB, 0, stream>>>(Wt, WtT, FDIM, HID);
    cvt_transpose_bf16<<<(HID * HF + TB - 1) / TB, TB, 0, stream>>>(Wg[0], WgT0, HID, HF);
    cvt_transpose_bf16<<<(HID * HF + TB - 1) / TB, TB, 0, stream>>>(Wg[1], WgT1, HID, HF);
    cvt_transpose_bf16<<<(HID * EMB + TB - 1) / TB, TB, 0, stream>>>(Wp, WpT, HID, EMB);

    // --- x = features @ Wt + bt (also emit bf16 mirror for next GEMM) ---
    gemm_bf16_wmma<<<dim3(MB, 1), 128, 0, stream>>>(feat_bf, WtT, bt, x, x_bf, FDIM, HID);

    // --- two GAT blocks ---
    for (int l = 0; l < 2; ++l) {
        gemm_bf16_wmma<<<dim3(MB, HF / 64), 128, 0, stream>>>(
            x_bf, WgT[l], nullptr, hbuf, h_bf, HID, HF);
        attn_logits<<<(NN * HEADS + TB - 1) / TB, TB, 0, stream>>>(hbuf, As[l], Ad[l], als, ald);
        fillf<<<(NN * HEADS + TB - 1) / TB, TB, 0, stream>>>(mbuf, -__builtin_huge_valf(), NN * HEADS);
        fillf<<<(NN * HEADS + TB - 1) / TB, TB, 0, stream>>>(denom, 0.f, NN * HEADS);
        fillf<<<(NN * HF + TB - 1) / TB, TB, 0, stream>>>(out_acc, 0.f, NN * HF);
        edge_max<<<(ETOT + TB - 1) / TB, TB, 0, stream>>>(ei, als, ald, mbuf);
        edge_sum<<<(ETOT + TB - 1) / TB, TB, 0, stream>>>(ei, als, ald, mbuf, denom);
        edge_agg<<<(ETOT + 7) / 8, TB, 0, stream>>>(ei, als, ald, mbuf, denom, h_bf, out_acc);
        node_finalize<<<(NN + 7) / 8, TB, 0, stream>>>(
            out_acc, Bg[l], Gm[l], Bb[l], Pa[l], x, x_bf);
    }

    // --- projection + L2 normalize ---
    gemm_bf16_wmma<<<dim3(MB, 1), 128, 0, stream>>>(x_bf, WpT, bp, zbuf, nullptr, HID, EMB);
    l2norm<<<(NN + 7) / 8, TB, 0, stream>>>(zbuf, (float*)d_out);
}
